// ReinmaxTernaryLinear_15058155339861
// MI455X (gfx1250) — matrix operand order repaired and checked by reading, not compiled, not run on hardware
//
#include <hip/hip_runtime.h>
#include <hip/hip_bf16.h>

// ---------------------------------------------------------------------------
// ReinmaxTernaryLinear on MI455X (gfx1250, wave32, WMMA, async + TDM):
//   Phase 1a: x[f32, M x K]        -> ws: Xb[bf16, M x K]      (row-major)
//   Phase 1b: logits/u/scales/mask -> ws: Wb[bf16, N x K]      (row-major)
//   Phase 2 : Out[f32, M x N] = Xb @ Wb^T via v_wmma_f32_16x16x32_bf16.
//     A tile: global_load_async_to_lds_b128 (ASYNCcnt, per-thread)
//     B tile: TENSOR_LOAD_TO_LDS, one TDM descriptor per K-step from wave 0
//             (TENSORcnt, s_wait_tensorcnt), double-buffered LDS.
// M = B*S = 8192, N = O = 4096, K = I = 4096.
// ws requirement: (M*K + N*K)*2 bytes = ~101 MB (fits entirely in 192MB L2).
// ---------------------------------------------------------------------------

typedef __attribute__((ext_vector_type(16))) __bf16 bf16x16;
typedef __attribute__((ext_vector_type(8)))  __bf16 bf16x8;
typedef __attribute__((ext_vector_type(4)))  __bf16 bf16x4;
typedef __attribute__((ext_vector_type(8)))  float  f32x8;
typedef __attribute__((ext_vector_type(4)))  float  f32x4;
typedef __attribute__((ext_vector_type(4)))  unsigned int v4u;
typedef __attribute__((ext_vector_type(8)))  int v8i;
typedef __attribute__((ext_vector_type(4)))  int v4i;

union ABFrag {
    bf16x16 v;
    struct { bf16x8 lo; bf16x8 hi; } h;
};

#define GROUP_SIZE 128
#define BM 128
#define BN 128
#define BK 32

#if __has_builtin(__builtin_amdgcn_tensor_load_to_lds)
#define HAVE_TDM 1
#else
#define HAVE_TDM 0
#endif

// Low 32 bits of a generic pointer to __shared__ == LDS byte offset
// (flat LDS address = {shared aperture, offset[31:0]}, ISA 10.2).
__device__ __forceinline__ unsigned lds_off(const void* p) {
    return (unsigned)(unsigned long long)p;
}

// Async DMA one 16B chunk per lane: LDS[ldsoff] = MEM[gaddr]  (ASYNCcnt)
__device__ __forceinline__ void async_copy_b128(unsigned ldsoff, const void* gaddr) {
    asm volatile("global_load_async_to_lds_b128 %0, %1, off"
                 :: "v"(ldsoff), "v"(gaddr)
                 : "memory");
}

__device__ __forceinline__ void wait_async() {
    asm volatile("s_wait_asynccnt 0" ::: "memory");
}

__device__ __forceinline__ void wait_tensor() {
#if __has_builtin(__builtin_amdgcn_s_wait_tensorcnt)
    __builtin_amdgcn_s_wait_tensorcnt(0);
#else
    asm volatile("s_wait_tensorcnt 0" ::: "memory");
#endif
}

#if HAVE_TDM
// TDM: load a 2D tile (tileD1 rows x tileD0 elements, 2-byte elements,
// row stride strideD0 elements) from global memory into LDS.
// D# packing per CDNA5 ISA 8.3/8.4; groups 2/3 zero (tile_dim2=0 => 2D).
__device__ __forceinline__ void tdm_load_2d_b16(unsigned lds_addr, const void* gaddr,
                                                unsigned tileD0, unsigned tileD1,
                                                unsigned tensorD0, unsigned tensorD1,
                                                unsigned strideD0) {
    unsigned long long ga = (unsigned long long)gaddr;
    v4u g0;
    g0[0] = 1u;                                          // count=1, user mode
    g0[1] = lds_addr;                                    // lds_addr[31:0]
    g0[2] = (unsigned)(ga & 0xffffffffu);                // global_addr[31:0]
    g0[3] = (unsigned)((ga >> 32) & 0x1ffffffu)          // global_addr[56:32]
          | (2u << 30);                                  // type = 2 ("image")
    v8i g1;
    g1[0] = (int)(1u << 16);                             // data_size=1 (2 bytes)
    g1[1] = (int)((tensorD0 & 0xffffu) << 16);           // tensor_dim0[15:0]
    g1[2] = (int)((tensorD0 >> 16) | ((tensorD1 & 0xffffu) << 16));
    g1[3] = (int)((tensorD1 >> 16) | ((tileD0 & 0xffffu) << 16));
    g1[4] = (int)(tileD1 & 0xffffu);                     // tile_dim1; tile_dim2=0
    g1[5] = (int)strideD0;                               // tensor_dim0_stride[31:0]
    g1[6] = 0;                                           // stride0 hi / stride1 lo
    g1[7] = 0;
    v4i z4 = {0, 0, 0, 0};
#if defined(__clang_major__) && (__clang_major__ >= 23)
    v8i z8 = {0, 0, 0, 0, 0, 0, 0, 0};
    __builtin_amdgcn_tensor_load_to_lds(g0, g1, z4, z4, z8, 0);
#else
    __builtin_amdgcn_tensor_load_to_lds(g0, g1, z4, z4, 0);
#endif
}
#endif

// ---------------------------------------------------------------------------
// Phase 1a: fp32 -> bf16 conversion of the activations (8 elems / thread)
// ---------------------------------------------------------------------------
__global__ __launch_bounds__(256) void cvt_x_kernel(const float* __restrict__ X,
                                                    __bf16* __restrict__ Xb,
                                                    long long total) {
    long long t = (long long)blockIdx.x * blockDim.x + threadIdx.x;
    long long base = t * 8;
    if (base >= total) return;
    f32x4 a = *(const f32x4*)(X + base);
    f32x4 b = *(const f32x4*)(X + base + 4);
    bf16x8 o;
    o[0] = (__bf16)a[0]; o[1] = (__bf16)a[1]; o[2] = (__bf16)a[2]; o[3] = (__bf16)a[3];
    o[4] = (__bf16)b[0]; o[5] = (__bf16)b[1]; o[6] = (__bf16)b[2]; o[7] = (__bf16)b[3];
    *(bf16x8*)(Xb + base) = o;
}

// ---------------------------------------------------------------------------
// Phase 1b: Gumbel argmax ternary sample + group dequant + mask -> bf16 W
// ---------------------------------------------------------------------------
__global__ __launch_bounds__(256) void gen_w_kernel(const float* __restrict__ logits,
                                                    const float* __restrict__ u,
                                                    const float* __restrict__ scales,
                                                    const int*   __restrict__ mask,
                                                    __bf16* __restrict__ Wb,
                                                    int O, int I) {
    long long t = (long long)blockIdx.x * blockDim.x + threadIdx.x;
    long long base = t * 4;
    if (base >= (long long)O * I) return;
    int o  = (int)(base / I);
    int i0 = (int)(base % I);
    float s = scales[(size_t)o * (I / GROUP_SIZE) + (i0 / GROUP_SIZE)];

    bf16x4 out;
    #pragma unroll
    for (int j = 0; j < 4; ++j) {
        long long e = base + j;
        const float* lp = logits + e * 3;
        const float* up = u + e * 3;
        float best = -__builtin_inff();
        int bi = 0;
        #pragma unroll
        for (int c = 0; c < 3; ++c) {
            float uu = up[c];
            float ex = -__logf(fmaxf(1.0f - uu, 1e-12f));
            float g  = -__logf(fmaxf(ex, 1e-10f));
            float sc = lp[c] + g;
            if (sc > best) { best = sc; bi = c; }
        }
        float w = (float)((bi == 2) - (bi == 0));
        w = w * s * (float)mask[e];
        out[j] = (__bf16)w;
    }
    *(bf16x4*)(Wb + base) = out;
}

// ---------------------------------------------------------------------------
// Phase 2: bf16 WMMA GEMM, A via async-LDS, B via TDM, double-buffered.
// Block = 128x128 output tile, 256 threads = 8 waves (wave32).
// Wave (wm in 0..1, wn in 0..3) owns 64x32 = 4x2 WMMA accumulators.
// ---------------------------------------------------------------------------
__global__ __launch_bounds__(256) void wmma_gemm_kernel(const __bf16* __restrict__ Xb,
                                                        const __bf16* __restrict__ Wb,
                                                        float* __restrict__ Out,
                                                        int M, int N, int K) {
    __shared__ __align__(16) __bf16 sA[2][BM * BK];   // 2 x 8 KB
    __shared__ __align__(16) __bf16 sB[2][BN * BK];   // 2 x 8 KB

    const int tid  = threadIdx.x;
    const int wave = tid >> 5;
    const int lane = tid & 31;
    const int half = lane >> 4;
    const int lm   = lane & 15;
    const int wm   = wave >> 2;   // 0..1
    const int wn   = wave & 3;    // 0..3

    const int bm0 = blockIdx.x * BM;
    const int bn0 = blockIdx.y * BN;

    // A staging: 128 rows x 4 chunks of 8 bf16 (16B); thread t handles
    // chunk t (row t/4) and chunk t+256 (row t/4 + 64).
    const int rS = tid >> 2;              // 0..63
    const int cS = (tid & 3) * 8;
    const size_t strideK = (size_t)K;

    const __bf16* gA0 = Xb + (size_t)(bm0 + rS) * strideK + cS;
    const __bf16* gA1 = gA0 + (size_t)64 * strideK;
#if !HAVE_TDM
    const __bf16* gB0 = Wb + (size_t)(bn0 + rS) * strideK + cS;
    const __bf16* gB1 = gB0 + (size_t)64 * strideK;
#endif
    const __bf16* gBtile = Wb + (size_t)bn0 * strideK;   // + kb at issue time

    const unsigned lA0 = lds_off(&sA[0][rS * BK + cS]);
    const unsigned lB0 = lds_off(&sB[0][rS * BK + cS]);
    const unsigned lBbase   = lds_off(&sB[0][0]);
    const unsigned bufStepA = (unsigned)(BM * BK * sizeof(__bf16));   // 8 KB
    const unsigned bufStepB = (unsigned)(BN * BK * sizeof(__bf16));   // 8 KB
    const unsigned rowStep  = (unsigned)(64 * BK * sizeof(__bf16));

    auto stage = [&](int buf, int kb) {
        unsigned la = lA0 + (unsigned)buf * bufStepA;
        async_copy_b128(la,           gA0 + kb);
        async_copy_b128(la + rowStep, gA1 + kb);
#if HAVE_TDM
        if (wave == 0) {
            // One TDM descriptor moves the whole 128x32 bf16 B tile.
            tdm_load_2d_b16(lBbase + (unsigned)buf * bufStepB,
                            gBtile + kb,
                            /*tileD0=*/BK, /*tileD1=*/BN,
                            /*tensorD0=*/(unsigned)K, /*tensorD1=*/(unsigned)N,
                            /*strideD0=*/(unsigned)K);
        }
#else
        unsigned lb = lB0 + (unsigned)buf * bufStepB;
        async_copy_b128(lb,           gB0 + kb);
        async_copy_b128(lb + rowStep, gB1 + kb);
#endif
    };

    f32x8 acc[4][2] = {};

    // Prologue: stage K-step 0 into buffer 0.
    stage(0, 0);
    wait_async();
    wait_tensor();
    __syncthreads();

    int cur = 0;
    for (int kb = 0; kb < K; kb += BK) {
        // Kick off next tile's DMA while we compute on the current one.
        if (kb + BK < K) stage(cur ^ 1, kb + BK);

        ABFrag a[4], b[2];
        #pragma unroll
        for (int mt = 0; mt < 4; ++mt) {
            const int row = wm * 64 + mt * 16 + lm;
            const __bf16* p = &sA[cur][row * BK + half * 8];
            a[mt].h.lo = *(const bf16x8*)(p);        // K[half*8 .. +7]
            a[mt].h.hi = *(const bf16x8*)(p + 16);   // K[16+half*8 .. +7]
        }
        #pragma unroll
        for (int nt = 0; nt < 2; ++nt) {
            const int row = wn * 32 + nt * 16 + lm;
            const __bf16* p = &sB[cur][row * BK + half * 8];
            b[nt].h.lo = *(const bf16x8*)(p);        // K[half*8 .. +7]
            b[nt].h.hi = *(const bf16x8*)(p + 16);   // K[16+half*8 .. +7]
        }

        #pragma unroll
        for (int mt = 0; mt < 4; ++mt)
            #pragma unroll
            for (int nt = 0; nt < 2; ++nt)
                acc[mt][nt] = __builtin_amdgcn_wmma_f32_16x16x32_bf16(
                    false, a[mt].v, false, b[nt].v,
                    (short)0, acc[mt][nt], false, false);

        // All waves done reading 'cur'; next buffer's DMAs complete.
        wait_async();
        wait_tensor();
        __syncthreads();
        cur ^= 1;
    }

    // Store: row = bm0 + wm*64 + mt*16 + half*8 + v, col = bn0 + wn*32 + nt*16 + lm
    #pragma unroll
    for (int mt = 0; mt < 4; ++mt) {
        #pragma unroll
        for (int nt = 0; nt < 2; ++nt) {
            float* outp = Out + (size_t)(bm0 + wm * 64 + mt * 16 + half * 8) * N
                              + (bn0 + wn * 32 + nt * 16 + lm);
            #pragma unroll
            for (int v = 0; v < 8; ++v)
                outp[(size_t)v * N] = acc[mt][nt][v];
        }
    }
}

// ---------------------------------------------------------------------------
extern "C" void kernel_launch(void* const* d_in, const int* in_sizes, int n_in,
                              void* d_out, int out_size, void* d_ws, size_t ws_size,
                              hipStream_t stream) {
    const float* x      = (const float*)d_in[0];   // [4, 2048, 4096]
    const float* logits = (const float*)d_in[1];   // [4096, 4096, 3]
    const float* u      = (const float*)d_in[2];   // [4096, 4096, 3]
    const float* scales = (const float*)d_in[3];   // [4096, 32]
    const int*   mask   = (const int*)d_in[4];     // [4096, 4096]
    float* out = (float*)d_out;                    // [4, 2048, 4096]

    const int B = 4, S = 2048, I = 4096, O = 4096;
    const int M = B * S;   // 8192
    const int N = O;       // 4096
    const int K = I;       // 4096

    __bf16* Xb = (__bf16*)d_ws;                 // M*K bf16 = 64 MB
    __bf16* Wb = Xb + (size_t)M * K;            // N*K bf16 = 32 MB

    // Phase 1a: activations fp32 -> bf16
    {
        long long total = (long long)M * K;
        int blocks = (int)(total / 8 / 256);
        cvt_x_kernel<<<blocks, 256, 0, stream>>>(x, Xb, total);
    }
    // Phase 1b: ternary sample + dequant + mask -> bf16 weights
    {
        long long total = (long long)O * I;
        int blocks = (int)(total / 4 / 256);
        gen_w_kernel<<<blocks, 256, 0, stream>>>(logits, u, scales, mask, Wb, O, I);
    }
    // Phase 2: WMMA GEMM with async-LDS (A) + TDM (B) double buffering
    {
        dim3 grid(M / BM, N / BN);                       // 64 x 32
        wmma_gemm_kernel<<<grid, 256, 0, stream>>>(Xb, Wb, out, M, N, K);
    }
}
